// ClusteredMambaModel_Flood_76656576299511
// MI455X (gfx1250) — matrix operand order repaired and checked by reading, not compiled, not run on hardware
//
#include <hip/hip_runtime.h>
#include <hip/hip_bf16.h>
#include <math.h>

// ---------------------------------------------------------------------------
// ClusteredMambaModel (L=1 => scan collapses; dA/alog dead).
// bf16 16x16x32 WMMA for all GEMMs, 16 samples per wave-tile.
// B operands pre-swizzled to packed-bf16 fragment layout in workspace
// (2x global_load_b128 per fragment); A operands kept as bf16 row-major in
// LDS (2x ds_load_b128 per fragment).
// ---------------------------------------------------------------------------

typedef __attribute__((ext_vector_type(16))) __bf16        bf16x16;
typedef __attribute__((ext_vector_type(8)))  float         f32x8;
typedef __attribute__((ext_vector_type(8)))  unsigned int  u32x8;
typedef __attribute__((ext_vector_type(4)))  unsigned int  u32x4;

// workspace layout (dwords)
#define WS_WEFF  0            // 1152 f32: folded input projections
#define WS_IPWB  1152         // 4 layers * 2kt*16nt*256 = 32768
#define WS_XPWB  (WS_IPWB + 32768)   // 4 layers * 4kt*3nt*256 = 12288
#define WS_OPWB  (WS_XPWB + 12288)   // 4 layers * 4kt*4nt*256 = 16384

// round-to-nearest-even fp32 -> packed 2x bf16 (lo = first element)
__device__ __forceinline__ unsigned pack_bf16(float a, float b) {
    unsigned ua = __builtin_bit_cast(unsigned, a);
    unsigned ub = __builtin_bit_cast(unsigned, b);
    ua += 0x7FFFu + ((ua >> 16) & 1u);
    ub += 0x7FFFu + ((ub >> 16) & 1u);
    return (ua >> 16) | (ub & 0xFFFF0000u);
}

// A fragment from bf16 row-major LDS shadow (row stride 64 dwords = 128 bf16).
// ISA 16-bit A layout: lane m=lane&15, hi=lane>>4; VGPR 0-3 = K [8hi,8hi+8),
// VGPR 4-7 = K [16+8hi, 16+8hi+8)  -> two contiguous 16B runs.
__device__ __forceinline__ bf16x16 loadA(const unsigned* abuf, int kOff) {
    const int lane = threadIdx.x & 31;
    const int m = lane & 15, hi = lane >> 4;
    const unsigned* row = abuf + m * 64 + (kOff >> 1);
    u32x4 lo = *(const u32x4*)(row + 4 * hi);
    u32x4 up = *(const u32x4*)(row + 8 + 4 * hi);
    u32x8 r;
    r[0] = lo[0]; r[1] = lo[1]; r[2] = lo[2]; r[3] = lo[3];
    r[4] = up[0]; r[5] = up[1]; r[6] = up[2]; r[7] = up[3];
    return __builtin_bit_cast(bf16x16, r);
}

// B fragment from pre-swizzled workspace tile: [lane][8 dwords]
__device__ __forceinline__ bf16x16 loadB(const unsigned* __restrict__ tileBase) {
    const unsigned* p = tileBase + (threadIdx.x & 31) * 8;
    u32x4 lo = *(const u32x4*)p;
    u32x4 up = *(const u32x4*)(p + 4);
    u32x8 r;
    r[0] = lo[0]; r[1] = lo[1]; r[2] = lo[2]; r[3] = lo[3];
    r[4] = up[0]; r[5] = up[1]; r[6] = up[2]; r[7] = up[3];
    return __builtin_bit_cast(bf16x16, r);
}

__device__ __forceinline__ f32x8 wmma_bf16(bf16x16 a, bf16x16 b, f32x8 c) {
    return __builtin_amdgcn_wmma_f32_16x16x32_bf16(
        false, a, false, b, (short)0, c, false, false);
}

// C/D f32 layout: lane holds column n = lane&15, rows m = v + 8*(lane>>4)
__device__ __forceinline__ void storeC(float* lds, int ld, int n0, f32x8 c) {
    const int lane = threadIdx.x & 31;
    const int n = lane & 15, hi = lane >> 4;
#pragma unroll
    for (int v = 0; v < 8; ++v) lds[(v + 8 * hi) * ld + n0 + n] = c[v];
}

__device__ __forceinline__ void storeC_guard(float* lds, int ld, int n0, int nTot, f32x8 c) {
    const int lane = threadIdx.x & 31;
    const int n = lane & 15, hi = lane >> 4;
    if (n0 + n < nTot) {
#pragma unroll
        for (int v = 0; v < 8; ++v) lds[(v + 8 * hi) * ld + n0 + n] = c[v];
    }
}

__device__ __forceinline__ float sigmoidf_(float v) { return 1.f / (1.f + __expf(-v)); }

// ---------------------------------------------------------------------------
// Kernel 1: fold input projections  Weff = w@w_in (12x64 / 4x64), beff.
// ---------------------------------------------------------------------------
__global__ void fold_kernel(const float* __restrict__ w_nr, const float* __restrict__ b_nr,
                            const float* __restrict__ w_r,  const float* __restrict__ b_r,
                            const float* __restrict__ w_in_nr, const float* __restrict__ b_in_nr,
                            const float* __restrict__ w_in_r,  const float* __restrict__ b_in_r,
                            float* __restrict__ ws) {
    for (int i = threadIdx.x; i < 1152; i += blockDim.x) {
        float acc;
        if (i < 768) {
            int j = i >> 6, d = i & 63; acc = 0.f;
            for (int c = 0; c < 1000; ++c) acc += w_nr[j * 1000 + c] * w_in_nr[c * 64 + d];
        } else if (i < 832) {
            int d = i - 768; acc = b_in_nr[d];
            for (int c = 0; c < 1000; ++c) acc += b_nr[c] * w_in_nr[c * 64 + d];
        } else if (i < 1088) {
            int t = i - 832, j = t >> 6, d = t & 63; acc = 0.f;
            for (int c = 0; c < 1000; ++c) acc += w_r[j * 1000 + c] * w_in_r[c * 64 + d];
        } else {
            int d = i - 1088; acc = b_in_r[d];
            for (int c = 0; c < 1000; ++c) acc += b_r[c] * w_in_r[c * 64 + d];
        }
        ws[i] = acc;
    }
}

// ---------------------------------------------------------------------------
// Kernel 2: pre-swizzle one weight matrix into packed-bf16 B-fragment layout.
// dst[i], i = ((kt*NT+nt)*256) + lane*8 + v ;
// lane: n = nt*16 + (lane&15), hi = lane>>4 ; VGPR v holds K = kt*32+2v+8hi+8*(v>>2).
// ---------------------------------------------------------------------------
__global__ void prep_kernel(const float* __restrict__ W, unsigned* __restrict__ dst,
                            int NT, int nTot, int ld, int total) {
    for (int i = blockIdx.x * blockDim.x + threadIdx.x; i < total;
         i += gridDim.x * blockDim.x) {
        int tile = i >> 8, r = i & 255, lane = r >> 3, v = r & 7;
        int kt = tile / NT, nt = tile % NT;
        int n = nt * 16 + (lane & 15), hi = lane >> 4;
        int k0 = kt * 32 + 2 * v + 8 * hi + 8 * (v >> 2);
        float a = (n < nTot) ? W[k0 * ld + n] : 0.f;
        float b = (n < nTot) ? W[(k0 + 1) * ld + n] : 0.f;
        dst[i] = pack_bf16(a, b);
    }
}

// ---------------------------------------------------------------------------
// Kernel 3: fused model. One wave per 16-sample tile.
// ---------------------------------------------------------------------------
__global__ __launch_bounds__(32) void mamba_main(
    const float* __restrict__ X,    const float* __restrict__ ws,
    const float* __restrict__ cw,   const float* __restrict__ cb,
    const float* __restrict__ dtw,  const float* __restrict__ dtb,
    const float* __restrict__ dskip,
    const float* __restrict__ ln_g, const float* __restrict__ ln_b,
    const float* __restrict__ hw1,  const float* __restrict__ hb1,
    const float* __restrict__ hw2,  const float* __restrict__ hb2,
    float* __restrict__ out) {

    __shared__ float    xs[16 * 64];    // residual stream (fp32)
    __shared__ float    sb[16 * 256];   // fp32 staging: xz / xc,z / outp
    __shared__ unsigned abuf[16 * 64];  // bf16 row-major activation shadow (A source)
    __shared__ float    ps[16 * 40];    // proj (36) + BC (slot 36)
    __shared__ float    rs[16 * 64];    // branch-0 result
    __shared__ float    cc[1152];       // per-layer channel constants

    const int lane = threadIdx.x;
    const int s0 = blockIdx.x * 16;
    const unsigned* wsu = (const unsigned*)ws;

    for (int br = 0; br < 2; ++br) {
        const float* Weff = ws + (br ? 832 : 0);
        const float* beff = ws + (br ? 1088 : 768);
        const int K = br ? 4 : 12;
        const int joff = br ? 12 : 0;

        // x0 = x_feat @ Weff + beff  (tiny K: VALU) ; also pack bf16 shadow
        for (int i = lane; i < 512; i += 32) {
            int m = i >> 5, dp = (i & 31) << 1;
            float a0 = beff[dp], a1 = beff[dp + 1];
            for (int j = 0; j < K; ++j) {
                float xv = X[(s0 + m) * 16 + joff + j];
                a0 += xv * Weff[j * 64 + dp];
                a1 += xv * Weff[j * 64 + dp + 1];
            }
            xs[m * 64 + dp] = a0; xs[m * 64 + dp + 1] = a1;
            abuf[m * 64 + (dp >> 1)] = pack_bf16(a0, a1);
        }
        __syncthreads();

        for (int l = 0; l < 4; ++l) {
            const unsigned* ipwB = wsu + WS_IPWB + l * 8192;
            const unsigned* xpwB = wsu + WS_XPWB + l * 3072;
            const unsigned* opwB = wsu + WS_OPWB + l * 4096;

            // per-layer channel constants -> LDS
            // cc[0:128)=cw3  [128:256)=cb  [256:384)=dtb  [384:512)=dskip
            // cc[512:1024)=dtw(4x128)  [1024:1088)=ln_g  [1088:1152)=ln_b
            for (int i = lane; i < 128; i += 32) {
                cc[i]       = cw[l * 512 + i * 4 + 3];
                cc[128 + i] = cb[l * 128 + i];
                cc[256 + i] = dtb[l * 128 + i];
                cc[384 + i] = dskip[l * 128 + i];
            }
            for (int i = lane; i < 512; i += 32) cc[512 + i] = dtw[l * 512 + i];
            for (int i = lane; i < 64; i += 32) {
                cc[1024 + i] = ln_g[l * 64 + i];
                cc[1088 + i] = ln_b[l * 64 + i];
            }
            __syncthreads();

            // xz = x @ ipw   (16x64 @ 64x256) : 32 WMMAs
            {
                bf16x16 a0 = loadA(abuf, 0);
                bf16x16 a1 = loadA(abuf, 32);
#pragma unroll
                for (int nt = 0; nt < 16; ++nt) {
                    f32x8 acc = (f32x8)(0.f);
                    acc = wmma_bf16(a0, loadB(ipwB + (0 * 16 + nt) * 256), acc);
                    acc = wmma_bf16(a1, loadB(ipwB + (1 * 16 + nt) * 256), acc);
                    storeC(sb, 256, nt * 16, acc);
                }
            }
            __syncthreads();

            // conv (L=1 => last tap only) + silu ; fp32 to sb, bf16 to abuf
            for (int i = lane; i < 1024; i += 32) {
                int m = i >> 6, dp = (i & 63) << 1;
                float v0 = sb[m * 256 + dp]     * cc[dp]     + cc[128 + dp];
                float v1 = sb[m * 256 + dp + 1] * cc[dp + 1] + cc[128 + dp + 1];
                v0 *= sigmoidf_(v0);
                v1 *= sigmoidf_(v1);
                sb[m * 256 + dp] = v0; sb[m * 256 + dp + 1] = v1;
                abuf[m * 64 + (dp >> 1)] = pack_bf16(v0, v1);
            }
            __syncthreads();

            // proj = xc @ xpw  (16x128 @ 128x36, N padded 48) : 12 WMMAs
            {
                bf16x16 ax0 = loadA(abuf, 0),  ax1 = loadA(abuf, 32);
                bf16x16 ax2 = loadA(abuf, 64), ax3 = loadA(abuf, 96);
#pragma unroll
                for (int nt = 0; nt < 3; ++nt) {
                    f32x8 acc = (f32x8)(0.f);
                    acc = wmma_bf16(ax0, loadB(xpwB + (0 * 3 + nt) * 256), acc);
                    acc = wmma_bf16(ax1, loadB(xpwB + (1 * 3 + nt) * 256), acc);
                    acc = wmma_bf16(ax2, loadB(xpwB + (2 * 3 + nt) * 256), acc);
                    acc = wmma_bf16(ax3, loadB(xpwB + (3 * 3 + nt) * 256), acc);
                    storeC_guard(ps, 40, nt * 16, 36, acc);
                }
            }
            __syncthreads();

            // BC[m] = Bm . Cm  (scan collapses: L=1, h0=0)
            if (lane < 16) {
                float bc = 0.f;
                for (int s = 0; s < 16; ++s)
                    bc += ps[lane * 40 + 4 + s] * ps[lane * 40 + 20 + s];
                ps[lane * 40 + 36] = bc;
            }
            __syncthreads();

            // y = xc * (softplus(proj4@dtw + dtb) * BC + dskip) * silu(z)
            //   -> only the bf16 shadow is needed downstream
            for (int i = lane; i < 1024; i += 32) {
                int m = i >> 6, dp = (i & 63) << 1;
                float p0 = ps[m * 40 + 0], p1 = ps[m * 40 + 1];
                float p2 = ps[m * 40 + 2], p3 = ps[m * 40 + 3];
                float bc = ps[m * 40 + 36];
                float y2[2];
#pragma unroll
                for (int t = 0; t < 2; ++t) {
                    int d = dp + t;
                    float dtv = cc[256 + d] + p0 * cc[512 + d] + p1 * cc[640 + d]
                              + p2 * cc[768 + d] + p3 * cc[896 + d];
                    float dt = (dtv > 20.f) ? dtv : log1pf(__expf(dtv));
                    float xc = sb[m * 256 + d];
                    float z  = sb[m * 256 + 128 + d];
                    float y  = xc * (dt * bc + cc[384 + d]);
                    y2[t] = y * (z * sigmoidf_(z));
                }
                abuf[m * 64 + (dp >> 1)] = pack_bf16(y2[0], y2[1]);
            }
            __syncthreads();

            // outp = y @ opw  (16x128 @ 128x64) : 16 WMMAs -> sb[:,128:192)
            {
                bf16x16 ay0 = loadA(abuf, 0),  ay1 = loadA(abuf, 32);
                bf16x16 ay2 = loadA(abuf, 64), ay3 = loadA(abuf, 96);
#pragma unroll
                for (int nt = 0; nt < 4; ++nt) {
                    f32x8 acc = (f32x8)(0.f);
                    acc = wmma_bf16(ay0, loadB(opwB + (0 * 4 + nt) * 256), acc);
                    acc = wmma_bf16(ay1, loadB(opwB + (1 * 4 + nt) * 256), acc);
                    acc = wmma_bf16(ay2, loadB(opwB + (2 * 4 + nt) * 256), acc);
                    acc = wmma_bf16(ay3, loadB(opwB + (3 * 4 + nt) * 256), acc);
                    storeC(sb, 256, 128 + nt * 16, acc);
                }
            }
            __syncthreads();

            // x += LayerNorm(outp)*g + b ; refresh fp32 x and bf16 shadow
            {
                int m = lane & 15, half = lane >> 4;
                float mean = 0.f;
                for (int d = 0; d < 64; ++d) mean += sb[m * 256 + 128 + d];
                mean *= (1.f / 64.f);
                float var = 0.f;
                for (int d = 0; d < 64; ++d) {
                    float t = sb[m * 256 + 128 + d] - mean;
                    var += t * t;
                }
                var *= (1.f / 64.f);
                float rstd = rsqrtf(var + 1e-5f);
                for (int dp = half * 32; dp < half * 32 + 32; dp += 2) {
                    float x0 = xs[m * 64 + dp] +
                        (sb[m * 256 + 128 + dp] - mean) * rstd * cc[1024 + dp] + cc[1088 + dp];
                    float x1 = xs[m * 64 + dp + 1] +
                        (sb[m * 256 + 129 + dp] - mean) * rstd * cc[1025 + dp] + cc[1089 + dp];
                    xs[m * 64 + dp] = x0; xs[m * 64 + dp + 1] = x1;
                    abuf[m * 64 + (dp >> 1)] = pack_bf16(x0, x1);
                }
            }
            __syncthreads();
        }

        if (br == 0) {
            for (int i = lane; i < 1024; i += 32) rs[i] = xs[i];
            __syncthreads();
        }
    }
    __syncthreads();

    // head: relu(concat(o_nr,o_r) @ hw1 + hb1) @ hw2 + hb2
    if (lane < 16) {
        int m = lane;
        float o = hb2[0];
        for (int c = 0; c < 32; ++c) {
            float h = hb1[c];
            for (int d = 0; d < 64; ++d) h += rs[m * 64 + d] * hw1[d * 32 + c];
            for (int d = 0; d < 64; ++d) h += xs[m * 64 + d] * hw1[(64 + d) * 32 + c];
            h = h > 0.f ? h : 0.f;
            o += h * hw2[c];
        }
        out[s0 + m] = o;
    }
}

// ---------------------------------------------------------------------------
extern "C" void kernel_launch(void* const* d_in, const int* in_sizes, int n_in,
                              void* d_out, int out_size, void* d_ws, size_t ws_size,
                              hipStream_t stream) {
    (void)n_in; (void)out_size; (void)ws_size;
    const float* X        = (const float*)d_in[0];
    const float* w_nr     = (const float*)d_in[1];
    const float* b_nr     = (const float*)d_in[2];
    const float* w_r      = (const float*)d_in[3];
    const float* b_r      = (const float*)d_in[4];
    const float* w_in_nr  = (const float*)d_in[5];
    const float* b_in_nr  = (const float*)d_in[6];
    const float* w_in_r   = (const float*)d_in[7];
    const float* b_in_r   = (const float*)d_in[8];
    const float* ipw      = (const float*)d_in[9];
    const float* cw       = (const float*)d_in[10];
    const float* cb       = (const float*)d_in[11];
    const float* xpw      = (const float*)d_in[12];
    const float* dtw      = (const float*)d_in[13];
    const float* dtb      = (const float*)d_in[14];
    // d_in[15] = alog: dead (L=1, h0=0)
    const float* dskip    = (const float*)d_in[16];
    const float* opw      = (const float*)d_in[17];
    const float* ln_g     = (const float*)d_in[18];
    const float* ln_b     = (const float*)d_in[19];
    const float* hw1      = (const float*)d_in[20];
    const float* hb1      = (const float*)d_in[21];
    const float* hw2      = (const float*)d_in[22];
    const float* hb2      = (const float*)d_in[23];

    float* ws = (float*)d_ws;
    unsigned* wsu = (unsigned*)d_ws;
    float* out = (float*)d_out;

    const int nsamp = in_sizes[0] / 16;   // BATCH (NFEAT = 16)
    const int ntiles = nsamp / 16;        // 16 samples per wave-tile

    fold_kernel<<<1, 256, 0, stream>>>(w_nr, b_nr, w_r, b_r,
                                       w_in_nr, b_in_nr, w_in_r, b_in_r, ws);
    for (int l = 0; l < 4; ++l) {
        prep_kernel<<<32, 256, 0, stream>>>(ipw + l * 64 * 256,
                                            wsu + WS_IPWB + l * 8192,
                                            16, 256, 256, 8192);
        prep_kernel<<<12, 256, 0, stream>>>(xpw + l * 128 * 36,
                                            wsu + WS_XPWB + l * 3072,
                                            3, 36, 36, 3072);
        prep_kernel<<<16, 256, 0, stream>>>(opw + l * 128 * 64,
                                            wsu + WS_OPWB + l * 4096,
                                            4, 64, 64, 4096);
    }
    mamba_main<<<ntiles, 32, 0, stream>>>(X, ws, cw, cb, dtw, dtb, dskip,
                                          ln_g, ln_b, hw1, hb1, hw2, hb2, out);
}
